// SparseWindowedAttention_15805479649987
// MI455X (gfx1250) — compile-verified, matrix operand order checked by reading, not run on hardware
//
#include <hip/hip_runtime.h>
#include <hip/hip_bf16.h>

#define S_LEN 4096
#define EMB   768
#define NH    12
#define HD    64
#define WIN   128

typedef _Float16 v16h __attribute__((ext_vector_type(16)));
typedef _Float16 v8h  __attribute__((ext_vector_type(8)));
typedef _Float16 v4h  __attribute__((ext_vector_type(4)));
typedef float    v8f  __attribute__((ext_vector_type(8)));

union V16U { v16h v; v8h h[2]; };

// Load a 16-lane-row x 32-K f16 WMMA A/B operand from a row-major source.
// Lane L holds row (passed in), halves e: K = k0 + 8*(L>>4) + e (e<8),
//                                         K = k0 + 16 + 8*(L>>4) + (e-8) (e>=8).
__device__ __forceinline__ v16h ld_op16(const _Float16* __restrict__ base,
                                        int row, int ld, int k0, int lane) {
  const _Float16* p = base + (size_t)row * ld + k0 + ((lane >> 4) << 3);
  V16U u;
  u.h[0] = *(const v8h*)(p);
  u.h[1] = *(const v8h*)(p + 16);
  return u.v;
}

__device__ __forceinline__ v8f wmma_f16(v16h a, v16h b, v8f c) {
  return __builtin_amdgcn_wmma_f32_16x16x32_f16(false, a, false, b, (short)0, c,
                                                false, false);
}

// ---------------------------------------------------------------- converts
// Vectorized: 4 floats -> 4 halves per thread (n must be a multiple of 4).
__global__ __launch_bounds__(256) void cvt_f32_to_f16_kernel(
    const float* __restrict__ in, _Float16* __restrict__ out, int n4) {
  int i = blockIdx.x * blockDim.x + threadIdx.x;
  if (i < n4) {
    const float4 f = ((const float4*)in)[i];
    v4h hlf;
    hlf.x = (_Float16)f.x; hlf.y = (_Float16)f.y;
    hlf.z = (_Float16)f.z; hlf.w = (_Float16)f.w;
    ((v4h*)out)[i] = hlf;
  }
}

// ------------------------------------------------------- pipelined operands
struct GemmOps { v16h a, b0, b1, b2, b3; };

__device__ __forceinline__ GemmOps ld_chunk(const _Float16* __restrict__ A,
                                            const _Float16* __restrict__ B,
                                            int mrow, int n0, int ln, int lane,
                                            int k0) {
  GemmOps t;
  t.a  = ld_op16(A, mrow,         EMB, k0, lane);
  t.b0 = ld_op16(B, n0 + ln,      EMB, k0, lane);
  t.b1 = ld_op16(B, n0 + 16 + ln, EMB, k0, lane);
  t.b2 = ld_op16(B, n0 + 32 + ln, EMB, k0, lane);
  t.b3 = ld_op16(B, n0 + 48 + ln, EMB, k0, lane);
  return t;
}

// ---------------------------------------------------------------- QKV GEMM
// qkv[s, n] = x[s,:] . Wqkv[n,:] + bqkv[n],  n = h*192 + t*64 + d
// t=0 -> Q (scaled 0.125, [h][s][d]), t=1 -> K ([h][s][d]), t=2 -> V^T ([h][d][s])
// Wave mapping: all 8 waves of a block share one 64-col B panel (WGP$ reuse).
__global__ __launch_bounds__(256) void qkv_gemm_kernel(
    const _Float16* __restrict__ xh, const _Float16* __restrict__ wh,
    const float* __restrict__ bqkv,
    _Float16* __restrict__ qh, _Float16* __restrict__ kh,
    _Float16* __restrict__ vTh) {
  const int lane = threadIdx.x & 31;
  const int wave = blockIdx.x * (blockDim.x >> 5) + (threadIdx.x >> 5);
  const int nb = wave >> 8;          // 0..35 : N-block (one (h,t) group each)
  const int mt = wave & 255;         // 0..255: M-tile (consecutive within block)
  const int m0 = mt * 16;
  const int n0 = nb * 64;
  const int ln = lane & 15;
  const int c  = lane >> 4;
  const int mrow = m0 + ln;

  v8f a0 = {0,0,0,0,0,0,0,0}, a1 = a0, a2 = a0, a3 = a0;
  GemmOps cur = ld_chunk(xh, wh, mrow, n0, ln, lane, 0);
  for (int k0 = 0; k0 < EMB; k0 += 32) {
    const int kn = (k0 + 32 < EMB) ? (k0 + 32) : k0;   // clamp: last iter reloads
    GemmOps nxt = ld_chunk(xh, wh, mrow, n0, ln, lane, kn);
    a0 = wmma_f16(cur.a, cur.b0, a0);
    a1 = wmma_f16(cur.a, cur.b1, a1);
    a2 = wmma_f16(cur.a, cur.b2, a2);
    a3 = wmma_f16(cur.a, cur.b3, a3);
    cur = nxt;
  }

  const int h  = nb / 3;
  const int tv = nb % 3;
  #pragma unroll
  for (int t = 0; t < 4; ++t) {
    v8f acc = (t == 0) ? a0 : (t == 1) ? a1 : (t == 2) ? a2 : a3;
    const int d = t * 16 + ln;           // 0..63 within head
    const float bias = bqkv[n0 + d];
    #pragma unroll
    for (int r = 0; r < 8; ++r) {
      const int s = m0 + r + 8 * c;
      const float val = acc[r] + bias;
      if (tv == 0)
        qh[((size_t)h * S_LEN + s) * HD + d] = (_Float16)(val * 0.125f);
      else if (tv == 1)
        kh[((size_t)h * S_LEN + s) * HD + d] = (_Float16)val;
      else
        vTh[((size_t)h * HD + d) * S_LEN + s] = (_Float16)val;
    }
  }
}

// ---------------------------------------------------------------- attention
// One wave = one head x one 16-query tile; flash-style over 32-key chunks.
// Pipelined: V loads and next-chunk K loads are issued before the softmax
// VALU sequence so they are in flight during it.
__global__ __launch_bounds__(128) void attn_kernel(
    const _Float16* __restrict__ qh, const _Float16* __restrict__ kh,
    const _Float16* __restrict__ vTh, _Float16* __restrict__ attnh) {
  __shared__ _Float16 plds[4][16 * 32];   // per-wave 16x32 P staging (1 KB)

  const int lane  = threadIdx.x & 31;
  const int wslot = threadIdx.x >> 5;
  const int wave  = blockIdx.x * 4 + wslot;   // 0..3071
  const int h  = wave >> 8;                   // / 256
  const int qt = wave & 255;                  // consecutive tiles share window
  const int i0 = qt * 16;
  const int ln = lane & 15;
  const int c  = lane >> 4;

  const _Float16* Qb = qh  + (size_t)h * S_LEN * HD;
  const _Float16* Kb = kh  + (size_t)h * S_LEN * HD;
  const _Float16* Vb = vTh + (size_t)h * HD * S_LEN;

  // Q A-operands (already scaled by 1/sqrt(hd)) for d-chunks 0 and 32
  const v16h qa0 = ld_op16(Qb, i0 + ln, HD, 0, lane);
  const v16h qa1 = ld_op16(Qb, i0 + ln, HD, 32, lane);

  v8f o0 = {0,0,0,0,0,0,0,0}, o1 = o0, o2 = o0, o3 = o0;
  float mrun[8], lrun[8];
  #pragma unroll
  for (int r = 0; r < 8; ++r) { mrun[r] = -__builtin_inff(); lrun[r] = 0.f; }

  int lo = i0 - WIN; if (lo < 0) lo = 0;
  int hi = i0 + 16 + WIN; if (hi > S_LEN) hi = S_LEN;
  const int jfirst = lo & ~31;
  _Float16* myp = plds[wslot];

  // K operands for the first chunk (two key tiles x two d-chunks)
  v16h kb0 = ld_op16(Kb, jfirst + ln,      HD, 0,  lane);
  v16h kb1 = ld_op16(Kb, jfirst + ln,      HD, 32, lane);
  v16h kb2 = ld_op16(Kb, jfirst + 16 + ln, HD, 0,  lane);
  v16h kb3 = ld_op16(Kb, jfirst + 16 + ln, HD, 32, lane);

  for (int j0 = jfirst; j0 < hi; j0 += 32) {
    // ---- scores: two 16x16 tiles over keys [j0, j0+32)
    v8f s0 = {0,0,0,0,0,0,0,0}, s1 = s0;
    s0 = wmma_f16(qa0, kb0, s0);
    s0 = wmma_f16(qa1, kb1, s0);
    s1 = wmma_f16(qa0, kb2, s1);
    s1 = wmma_f16(qa1, kb3, s1);

    // ---- issue V loads for this chunk now (independent of softmax)
    const v16h vb0 = ld_op16(Vb, ln,      S_LEN, j0, lane);
    const v16h vb1 = ld_op16(Vb, 16 + ln, S_LEN, j0, lane);
    const v16h vb2 = ld_op16(Vb, 32 + ln, S_LEN, j0, lane);
    const v16h vb3 = ld_op16(Vb, 48 + ln, S_LEN, j0, lane);

    // ---- issue next-chunk K loads (clamped; last iter reloads harmlessly)
    const int jn = (j0 + 32 < hi) ? (j0 + 32) : j0;
    kb0 = ld_op16(Kb, jn + ln,      HD, 0,  lane);
    kb1 = ld_op16(Kb, jn + ln,      HD, 32, lane);
    kb2 = ld_op16(Kb, jn + 16 + ln, HD, 0,  lane);
    kb3 = ld_op16(Kb, jn + 16 + ln, HD, 32, lane);

    // ---- mask + online softmax (row m = r + 8*c, keys across 16 lanes)
    const int kq0 = j0 + ln;
    const int kq1 = kq0 + 16;
    #pragma unroll
    for (int r = 0; r < 8; ++r) {
      const int q = i0 + r + 8 * c;
      const float NI = -__builtin_inff();
      float a = ((kq0 >= q - WIN) && (kq0 <= q + WIN)) ? s0[r] : NI;
      float b = ((kq1 >= q - WIN) && (kq1 <= q + WIN)) ? s1[r] : NI;
      float t = fmaxf(a, b);
      t = fmaxf(t, __shfl_xor(t, 1));
      t = fmaxf(t, __shfl_xor(t, 2));
      t = fmaxf(t, __shfl_xor(t, 4));
      t = fmaxf(t, __shfl_xor(t, 8));
      const float mnew  = fmaxf(mrun[r], t);
      const float alpha = __expf(mrun[r] - mnew);
      mrun[r] = mnew;
      const float p0 = __expf(a - mnew);
      const float p1 = __expf(b - mnew);
      float sum = p0 + p1;
      sum += __shfl_xor(sum, 1);
      sum += __shfl_xor(sum, 2);
      sum += __shfl_xor(sum, 4);
      sum += __shfl_xor(sum, 8);
      lrun[r] = lrun[r] * alpha + sum;
      o0[r] *= alpha; o1[r] *= alpha; o2[r] *= alpha; o3[r] *= alpha;
      // stage P (f16) into LDS in row-major [m][key_local]
      myp[(r + 8 * c) * 32 + ln]      = (_Float16)p0;
      myp[(r + 8 * c) * 32 + 16 + ln] = (_Float16)p1;
    }

    asm volatile("s_wait_dscnt 0x0" ::: "memory");

    // ---- reload P as A-operand (16 x 32 over key dim)
    const v16h pa = ld_op16(myp, ln, 32, 0, lane);

    // ---- O += P @ V  (4 hd-tiles; V operands already in flight)
    o0 = wmma_f16(pa, vb0, o0);
    o1 = wmma_f16(pa, vb1, o1);
    o2 = wmma_f16(pa, vb2, o2);
    o3 = wmma_f16(pa, vb3, o3);
  }

  // ---- normalize and write [s][h*64 + d] (f16 for the out-projection)
  #pragma unroll
  for (int r = 0; r < 8; ++r) {
    const int s = i0 + r + 8 * c;
    const float inv = 1.f / lrun[r];
    _Float16* dst = attnh + (size_t)s * EMB + h * HD + ln;
    dst[0]  = (_Float16)(o0[r] * inv);
    dst[16] = (_Float16)(o1[r] * inv);
    dst[32] = (_Float16)(o2[r] * inv);
    dst[48] = (_Float16)(o3[r] * inv);
  }
}

// ---------------------------------------------------------------- out GEMM
__global__ __launch_bounds__(256) void out_gemm_kernel(
    const _Float16* __restrict__ ah, const _Float16* __restrict__ woh,
    const float* __restrict__ bo, float* __restrict__ out) {
  const int lane = threadIdx.x & 31;
  const int wave = blockIdx.x * (blockDim.x >> 5) + (threadIdx.x >> 5);
  const int nb = wave >> 8;            // 0..11 : B panel shared by whole block
  const int mt = wave & 255;           // 0..255
  const int m0 = mt * 16;
  const int n0 = nb * 64;
  const int ln = lane & 15;
  const int c  = lane >> 4;
  const int mrow = m0 + ln;

  v8f a0 = {0,0,0,0,0,0,0,0}, a1 = a0, a2 = a0, a3 = a0;
  GemmOps cur = ld_chunk(ah, woh, mrow, n0, ln, lane, 0);
  for (int k0 = 0; k0 < EMB; k0 += 32) {
    const int kn = (k0 + 32 < EMB) ? (k0 + 32) : k0;
    GemmOps nxt = ld_chunk(ah, woh, mrow, n0, ln, lane, kn);
    a0 = wmma_f16(cur.a, cur.b0, a0);
    a1 = wmma_f16(cur.a, cur.b1, a1);
    a2 = wmma_f16(cur.a, cur.b2, a2);
    a3 = wmma_f16(cur.a, cur.b3, a3);
    cur = nxt;
  }

  #pragma unroll
  for (int t = 0; t < 4; ++t) {
    v8f acc = (t == 0) ? a0 : (t == 1) ? a1 : (t == 2) ? a2 : a3;
    const int ncol = n0 + t * 16 + ln;
    const float bias = bo[ncol];
    #pragma unroll
    for (int r = 0; r < 8; ++r) {
      const int s = m0 + r + 8 * c;
      out[(size_t)s * EMB + ncol] = acc[r] + bias;
    }
  }
}

// ---------------------------------------------------------------- launch
extern "C" void kernel_launch(void* const* d_in, const int* in_sizes, int n_in,
                              void* d_out, int out_size, void* d_ws, size_t ws_size,
                              hipStream_t stream) {
  (void)in_sizes; (void)n_in; (void)out_size; (void)ws_size;
  const float* x    = (const float*)d_in[0];
  const float* Wqkv = (const float*)d_in[1];
  const float* bqkv = (const float*)d_in[2];
  const float* Wo   = (const float*)d_in[3];
  const float* bo   = (const float*)d_in[4];
  // d_in[5] = window_size (==128, hardcoded to match reference setup)

  char* ws = (char*)d_ws;
  size_t off = 0;
  auto carve = [&](size_t bytes) {
    char* p = ws + off;
    off += (bytes + 255) & ~(size_t)255;
    return p;
  };
  _Float16* xh    = (_Float16*)carve((size_t)S_LEN * EMB * 2);       // also reused as attnh
  _Float16* wqkvh = (_Float16*)carve((size_t)3 * EMB * EMB * 2);
  _Float16* woh   = (_Float16*)carve((size_t)EMB * EMB * 2);
  _Float16* qh    = (_Float16*)carve((size_t)NH * S_LEN * HD * 2);
  _Float16* kh    = (_Float16*)carve((size_t)NH * S_LEN * HD * 2);
  _Float16* vTh   = (_Float16*)carve((size_t)NH * HD * S_LEN * 2);
  _Float16* attnh = xh;   // x_f16 dead after QKV GEMM -> reuse for attention out

  const int nx4 = (S_LEN * EMB) / 4;       // 786,432
  const int nw4 = (3 * EMB * EMB) / 4;     // 442,368
  const int no4 = (EMB * EMB) / 4;         // 147,456
  cvt_f32_to_f16_kernel<<<(nx4 + 255) / 256, 256, 0, stream>>>(x, xh, nx4);
  cvt_f32_to_f16_kernel<<<(nw4 + 255) / 256, 256, 0, stream>>>(Wqkv, wqkvh, nw4);
  cvt_f32_to_f16_kernel<<<(no4 + 255) / 256, 256, 0, stream>>>(Wo, woh, no4);

  // 36 N-blocks * 256 M-tiles = 9216 waves, 8 waves/block (block shares B panel)
  qkv_gemm_kernel<<<1152, 256, 0, stream>>>(xh, wqkvh, bqkv, qh, kh, vTh);

  // 12 heads * 256 query tiles = 3072 waves, 4 waves/block
  attn_kernel<<<768, 128, 0, stream>>>(qh, kh, vTh, attnh);

  // 12 N-blocks * 256 M-tiles = 3072 waves, 8 waves/block
  out_gemm_kernel<<<384, 256, 0, stream>>>(attnh, woh, bo, (float*)d_out);
}
